// PyTorchSimpleSNN_61916248539500
// MI455X (gfx1250) — matrix operand order repaired and checked by reading, not compiled, not run on hardware
//
#include <hip/hip_runtime.h>
#include <math.h>

// ---------------- types ----------------
typedef __bf16 bf16;
typedef bf16  v8bf  __attribute__((ext_vector_type(8)));
typedef bf16  v16bf __attribute__((ext_vector_type(16)));
typedef float v8f   __attribute__((ext_vector_type(8)));
typedef int   v4i   __attribute__((vector_size(16)));

// ---------------- problem constants ----------------
constexpr int    kN   = 4096;
constexpr size_t kMAT = (size_t)kN * kN;

// GEMM tiling: 128x128 block tile, K-step 32, 4 waves (128 threads),
// each wave computes a 64x64 sub-tile = 4x4 WMMA tiles of 16x16.
constexpr int kBM  = 128;
constexpr int kBN  = 128;
constexpr int kBK  = 32;
constexpr int kLDT = 40;   // padded LDS row stride in halves (80B, 16B aligned)

// ---------------- async global->LDS staging (CDNA5 path) ----------------
#if defined(__has_builtin)
#if __has_builtin(__builtin_amdgcn_global_load_async_to_lds_b128)
#define HAVE_ASYNC_LDS 1
#endif
#endif

static __device__ __forceinline__ void stage16(const bf16* g, bf16* l) {
#ifdef HAVE_ASYNC_LDS
    __builtin_amdgcn_global_load_async_to_lds_b128(
        (__attribute__((address_space(1))) v4i*)const_cast<bf16*>(g),
        (__attribute__((address_space(3))) v4i*)l, 0, 0);
#else
    *(v8bf*)l = *(const v8bf*)g;
#endif
}

static __device__ __forceinline__ void wait_async_prev() {
#ifdef HAVE_ASYNC_LDS
#if __has_builtin(__builtin_amdgcn_s_wait_asynccnt)
    __builtin_amdgcn_s_wait_asynccnt(8);   // 8 just-issued copies may stay in flight
#else
    asm volatile("s_wait_asynccnt 0x8" ::: "memory");
#endif
#endif
}

static __device__ __forceinline__ void wait_async_all() {
#ifdef HAVE_ASYNC_LDS
#if __has_builtin(__builtin_amdgcn_s_wait_asynccnt)
    __builtin_amdgcn_s_wait_asynccnt(0);
#else
    asm volatile("s_wait_asynccnt 0x0" ::: "memory");
#endif
#endif
}

// round-to-nearest-even fp32 -> bf16 bits (result in [31:16])
static __device__ __forceinline__ unsigned bfr(float f) {
    unsigned u = __float_as_uint(f);
    return u + 0x7FFFu + ((u >> 16) & 1u);
}

// ---------------- fp32 -> bf16 conversion (packed integer path) ----------------
__global__ __launch_bounds__(256)
void f32_to_bf16_kernel(const float* __restrict__ src, unsigned* __restrict__ dst, int n4) {
    int i = blockIdx.x * blockDim.x + threadIdx.x;
    int stride = gridDim.x * blockDim.x;
    for (; i < n4; i += stride) {
        float4 f = ((const float4*)src)[i];
        uint2 o;
        o.x = (bfr(f.x) >> 16) | (bfr(f.y) & 0xFFFF0000u);
        o.y = (bfr(f.z) >> 16) | (bfr(f.w) & 0xFFFF0000u);
        ((uint2*)dst)[i] = o;
    }
}

// ---------------- WMMA GEMM: C = A(BxK,row) * W(NxK,row)^T + bias ----------------
// EPI==0: write C fp32 + C bf16 (for GEMM2) + per-block partial sum of C
// EPI==1: write sigmoid(C) fp32
template <int EPI>
__global__ __launch_bounds__(128)
void gemm_bf16_wmma(const bf16* __restrict__ A, const bf16* __restrict__ W,
                    const float* __restrict__ bias,
                    float* __restrict__ Cf32, unsigned short* __restrict__ Cbf16,
                    float* __restrict__ psums) {
    __shared__ bf16 As[2][kBM * kLDT];
    __shared__ bf16 Bs[2][kBN * kLDT];

    const int tid  = threadIdx.x;          // 0..127
    const int lane = tid & 31;
    const int wv   = tid >> 5;             // 0..3
    const int wm   = wv >> 1;              // 0..1 (64-row band)
    const int wn   = wv & 1;               // 0..1 (64-col band)
    const int half = lane >> 4;            // 0..1
    const int l16  = lane & 15;

    const int m0 = blockIdx.y * kBM;
    const int n0 = blockIdx.x * kBN;

    v8f zero = {};
    v8f acc[4][4];
#pragma unroll
    for (int i = 0; i < 4; ++i)
#pragma unroll
        for (int j = 0; j < 4; ++j) acc[i][j] = zero;

    // stage one 128x32 tile of A and W into LDS buffer b (512 16B-chunks each, 4/thread)
    auto stage_tile = [&](int kk, int b) __attribute__((always_inline)) {
#pragma unroll
        for (int p = 0; p < 4; ++p) {
            int r = (p * 128 + tid) >> 2;
            int c = ((p * 128 + tid) & 3) * 8;
            stage16(&A[(size_t)(m0 + r) * kN + kk + c], &As[b][r * kLDT + c]);
            stage16(&W[(size_t)(n0 + r) * kN + kk + c], &Bs[b][r * kLDT + c]);
        }
    };

    // 16 WMMAs on LDS buffer b
    auto compute_tile = [&](int b) __attribute__((always_inline)) {
        // B fragments: lane holds W row n=l16 (of tile), k = half*16 .. +15 contiguous
        v16bf bfrag[4];
#pragma unroll
        for (int tn = 0; tn < 4; ++tn) {
            const bf16* bp = &Bs[b][(wn * 64 + tn * 16 + l16) * kLDT + half * 16];
            v8bf lo = *(const v8bf*)bp;
            v8bf hi = *(const v8bf*)(bp + 8);
            bfrag[tn] = __builtin_shufflevector(lo, hi, 0,1,2,3,4,5,6,7,8,9,10,11,12,13,14,15);
        }
        // A fragments: lane holds row m=l16, k = half*16 .. +15 contiguous
#pragma unroll
        for (int tm = 0; tm < 4; ++tm) {
            const bf16* ap = &As[b][(wm * 64 + tm * 16 + l16) * kLDT + half * 16];
            v8bf lo = *(const v8bf*)ap;
            v8bf hi = *(const v8bf*)(ap + 8);
            v16bf afrag = __builtin_shufflevector(lo, hi, 0,1,2,3,4,5,6,7,8,9,10,11,12,13,14,15);
#pragma unroll
            for (int tn = 0; tn < 4; ++tn) {
                acc[tm][tn] = __builtin_amdgcn_wmma_f32_16x16x32_bf16(
                    false, afrag, false, bfrag[tn], (short)0, acc[tm][tn], false, false);
            }
        }
    };

    // prologue: tile 0 in flight
    stage_tile(0, 0);

    // steady state: always stage next tile, retire previous, compute (last iter peeled)
    int buf = 0;
    for (int k0 = 0; k0 < kN - kBK; k0 += kBK) {
        stage_tile(k0 + kBK, buf ^ 1);
        wait_async_prev();
        __syncthreads();
        compute_tile(buf);
        __syncthreads();
        buf ^= 1;
    }
    // peeled final tile
    wait_async_all();
    __syncthreads();
    compute_tile(buf);

    // epilogue; C/D layout: VGPR r -> M = r + half*8, N = l16
    float psum = 0.0f;
#pragma unroll
    for (int tm = 0; tm < 4; ++tm) {
#pragma unroll
        for (int tn = 0; tn < 4; ++tn) {
            int col = n0 + wn * 64 + tn * 16 + l16;
            float bv = bias[col];
#pragma unroll
            for (int r = 0; r < 8; ++r) {
                int row = m0 + wm * 64 + tm * 16 + half * 8 + r;
                float v = acc[tm][tn][r] + bv;
                size_t idx = (size_t)row * kN + col;
                if (EPI == 0) {
                    Cf32[idx]  = v;
                    Cbf16[idx] = (unsigned short)(bfr(v) >> 16);
                    psum += v;
                } else {
                    Cf32[idx] = 1.0f / (1.0f + __expf(-v));
                }
            }
        }
    }
    if (EPI == 0) {
        __syncthreads();
        float* red = (float*)&As[0][0];
        red[tid] = psum;
        __syncthreads();
        for (int off = 64; off > 0; off >>= 1) {
            if (tid < off) red[tid] += red[tid + off];
            __syncthreads();
        }
        if (tid == 0) psums[blockIdx.y * gridDim.x + blockIdx.x] = red[0];
    }
}

// ---------------- scalar finalize (deterministic) ----------------
__global__ __launch_bounds__(256)
void finalize_scalars(const float* __restrict__ partials, int np,
                      const float* __restrict__ ga, float* __restrict__ scalars) {
    __shared__ float sm[256];
    float s = 0.0f;
    for (int i = threadIdx.x; i < np; i += 256) s += partials[i];
    sm[threadIdx.x] = s;
    __syncthreads();
    for (int off = 128; off > 0; off >>= 1) {
        if (threadIdx.x < off) sm[threadIdx.x] += sm[threadIdx.x + off];
        __syncthreads();
    }
    if (threadIdx.x == 0) {
        float activity = sm[0] / (float)kMAT;
        float thermal  = activity * 0.1f * 0.7f;       // DECAY_RATE
        float pos      = 1.0f + thermal * 0.2f;        // speedup == cycle position
        const float AT = 25.6f;                        // CYCLE_LENGTH * DUTY_RATIO
        float mod;
        if (pos < AT) mod = 0.5f + 0.5f * sinf(pos / AT * 3.14159265358979323846f);
        else          mod = 0.1f * expf(-3.0f * (pos - AT) / (32.0f - AT));
        float isact = (pos < AT) ? 1.0f : 0.0f;
        float fb    = 1.0f + activity * 0.2f * mod;    // FEEDBACK_GAIN
        scalars[0]  = mod * (1.0f + isact * (fb - 1.0f)); // combined prob scale
        scalars[1]  = ga[0] * (1.0f + mod);               // adaptive gain
    }
}

// ---------------- elementwise SNN epilogue ----------------
__global__ __launch_bounds__(256)
void snn_epilogue(float* __restrict__ out, const float* __restrict__ scalars, int n4) {
    const size_t M = kMAT;
    const float s    = scalars[0];
    const float gain = scalars[1];
    float4* spikes = (float4*)out;             // currently holds prob_weights
    float4* vmem   = (float4*)(out + M);
    float4* isyn   = (float4*)(out + 2 * M);   // currently holds x_processed
    int i = blockIdx.x * blockDim.x + threadIdx.x;
    int stride = gridDim.x * blockDim.x;
    for (; i < n4; i += stride) {
        float4 pw = spikes[i];
        float4 xp = isyn[i];
        float4 xm, vv, sp;
        float m0 = xp.x * (pw.x * s); float k0 = (m0 >= 1.0f) ? 1.0f : 0.0f;
        xm.x = m0; vv.x = m0 * (1.0f - k0); sp.x = k0 * gain;
        float m1 = xp.y * (pw.y * s); float k1 = (m1 >= 1.0f) ? 1.0f : 0.0f;
        xm.y = m1; vv.y = m1 * (1.0f - k1); sp.y = k1 * gain;
        float m2 = xp.z * (pw.z * s); float k2 = (m2 >= 1.0f) ? 1.0f : 0.0f;
        xm.z = m2; vv.z = m2 * (1.0f - k2); sp.z = k2 * gain;
        float m3 = xp.w * (pw.w * s); float k3 = (m3 >= 1.0f) ? 1.0f : 0.0f;
        xm.w = m3; vv.w = m3 * (1.0f - k3); sp.w = k3 * gain;
        spikes[i] = sp;
        vmem[i]   = vv;
        isyn[i]   = xm;
    }
}

// ---------------- launch ----------------
extern "C" void kernel_launch(void* const* d_in, const int* in_sizes, int n_in,
                              void* d_out, int out_size, void* d_ws, size_t ws_size,
                              hipStream_t stream) {
    const float* x      = (const float*)d_in[0];
    const float* W_in   = (const float*)d_in[1];
    const float* b_in   = (const float*)d_in[2];
    const float* W_gate = (const float*)d_in[3];
    const float* b_gate = (const float*)d_in[4];
    const float* ga     = (const float*)d_in[5];
    float* out = (float*)d_out;

    char* ws = (char*)d_ws;
    bf16*  xbf      = (bf16*)(ws);
    bf16*  winbf    = (bf16*)(ws + kMAT * 2);
    bf16*  wgbf     = (bf16*)(ws + kMAT * 4);
    bf16*  xpbf     = (bf16*)(ws + kMAT * 6);
    float* partials = (float*)(ws + kMAT * 8);
    float* scalars  = partials + 4096;

    float* xproc_f32 = out + 2 * kMAT;   // i_syn region used as scratch
    float* prob_f32  = out;              // spikes region used as scratch

    const int n4 = (int)(kMAT / 4);

    f32_to_bf16_kernel<<<4096, 256, 0, stream>>>(x,      (unsigned*)xbf,   n4);
    f32_to_bf16_kernel<<<4096, 256, 0, stream>>>(W_in,   (unsigned*)winbf, n4);
    f32_to_bf16_kernel<<<4096, 256, 0, stream>>>(W_gate, (unsigned*)wgbf,  n4);

    dim3 gg(kN / kBN, kN / kBM);   // 32 x 32 = 1024 blocks
    gemm_bf16_wmma<0><<<gg, 128, 0, stream>>>(xbf, winbf, b_in, xproc_f32,
                                              (unsigned short*)xpbf, partials);

    finalize_scalars<<<1, 256, 0, stream>>>(partials, (int)(gg.x * gg.y), ga, scalars);

    gemm_bf16_wmma<1><<<gg, 128, 0, stream>>>(xpbf, wgbf, b_gate, prob_f32,
                                              (unsigned short*)nullptr, (float*)nullptr);

    snn_epilogue<<<4096, 256, 0, stream>>>(out, scalars, n4);
}
